// NextRowPredictionHead_24455543783496
// MI455X (gfx1250) — matrix operand order repaired and checked by reading, not compiled, not run on hardware
//
#include <hip/hip_runtime.h>
#include <math.h>

// ---------------- types ----------------
typedef __bf16 bf16_t;
typedef bf16_t v16bf __attribute__((ext_vector_type(16)));
typedef float  v8f   __attribute__((ext_vector_type(8)));

static constexpr int kD  = 512;   // embedding dim
static constexpr int kH  = 256;   // head hidden
static constexpr int kNC = 32;    // categorical heads
static constexpr int kNN = 16;    // numeric heads
static constexpr int kNB = 16;    // bool heads
static constexpr int kNF = 64;    // total features
static constexpr int kV  = 1000;  // vocab
static constexpr int kVp = 1008;  // vocab padded to 16-col tiles
static constexpr int kB  = 2048;  // batch
static constexpr int kLdA = 280;  // LDS A-tile leading dim (16B-aligned, conflict-free)

#define NEG_BIG (-3.402823466e+38f)

// ---------------- WMMA fragment helpers (CDNA5, wave32) ----------------
union FragBF { v16bf v; uint4 q[2]; };

// A fragment: 16x32 bf16, row-major source with leading dim ld (elements).
__device__ __forceinline__ v16bf load_frag_a(const bf16_t* base, int ld) {
  const int lane = threadIdx.x & 31;
  const int r = lane & 15, hf = lane >> 4;
  const bf16_t* row = base + (size_t)r * ld;
  FragBF f;
  f.q[0] = *reinterpret_cast<const uint4*>(row + hf * 8);
  f.q[1] = *reinterpret_cast<const uint4*>(row + 16 + hf * 8);
  return f.v;
}

// B fragment: 32x16 bf16 from a column-major (transposed, [N][K]) source.
__device__ __forceinline__ v16bf load_frag_b(const bf16_t* baseT, int ld) {
  const int lane = threadIdx.x & 31;
  const int c = lane & 15, hf = lane >> 4;
  const bf16_t* p = baseT + (size_t)c * ld + hf * 8;
  FragBF f;
  f.q[0] = *reinterpret_cast<const uint4*>(p);
  f.q[1] = *reinterpret_cast<const uint4*>(p + 16);
  return f.v;
}

__device__ __forceinline__ v8f wmma_bf16(v16bf a, v16bf b, v8f c) {
  return __builtin_amdgcn_wmma_f32_16x16x32_bf16(false, a, false, b, (short)0, c,
                                                 false, false);
}

__device__ __forceinline__ v8f v8zero() {
  v8f z = {0.f, 0.f, 0.f, 0.f, 0.f, 0.f, 0.f, 0.f};
  return z;
}

// ---------------- kernel 0: weight convert/transpose to bf16, zero accum ----------------
__global__ void k_prep(const float* __restrict__ W1,  const float* __restrict__ Wc1,
                       const float* __restrict__ Wc2, const float* __restrict__ Wn1,
                       const float* __restrict__ Wn2, const float* __restrict__ Wb1,
                       const float* __restrict__ Wb2,
                       bf16_t* __restrict__ w1t,  bf16_t* __restrict__ wh1t,
                       bf16_t* __restrict__ wc2t, bf16_t* __restrict__ wn2t,
                       bf16_t* __restrict__ wb2t, float* __restrict__ loss_acc) {
  const int stride = gridDim.x * blockDim.x;
  const int tid0 = blockIdx.x * blockDim.x + threadIdx.x;

  for (int i = tid0; i < kD * kD; i += stride) {
    int e = i / kD, d = i % kD;
    w1t[i] = (bf16_t)W1[(size_t)d * kD + e];
  }
  for (int i = tid0; i < kNF * kH * kD; i += stride) {
    int f = i / (kH * kD), h = (i / kD) % kH, d = i % kD;
    float v;
    if (f < 32)      v = Wc1[((size_t)f * kD + d) * kH + h];
    else if (f < 48) v = Wn1[((size_t)(f - 32) * kD + d) * kH + h];
    else             v = Wb1[((size_t)(f - 48) * kD + d) * kH + h];
    wh1t[i] = (bf16_t)v;
  }
  for (int i = tid0; i < kNC * kVp * kH; i += stride) {
    int f = i / (kVp * kH), v = (i / kH) % kVp, k = i % kH;
    float x = (v < kV) ? Wc2[((size_t)f * kH + k) * kV + v] : 0.f;
    wc2t[i] = (bf16_t)x;
  }
  for (int i = tid0; i < kNN * kD * kH; i += stride) {
    int f = i / (kD * kH), d = (i / kH) % kD, k = i % kH;
    wn2t[i] = (bf16_t)Wn2[((size_t)f * kH + k) * kD + d];
  }
  for (int i = tid0; i < kNB * 2 * kH; i += stride) {
    int f = i / (2 * kH), c = (i / kH) % 2, k = i % kH;
    wb2t[i] = (bf16_t)Wb2[((size_t)f * kH + k) * 2 + c];
  }
  for (int i = tid0; i < kNF; i += stride) loss_acc[i] = 0.f;
}

// ---------------- kernel 1a: gelu(X @ W1 + b1) -> sh[f][b][d] (bf16, pre-LN) ----------
// Grid (2048, 4). Block 128 = 4 waves. Block tile 64 rows x 128 cols; wave = 64x32.
// __launch_bounds__(128, 1): lift the default waves-per-EU floor so the ~170-VGPR live
// set (64 acc + frags + staging) fits without accumulator spills.
__global__ void __launch_bounds__(128, 1)
k_gemm1(const float* __restrict__ seq, const bf16_t* __restrict__ w1t,
        const float* __restrict__ b1, bf16_t* __restrict__ sh) {
  __shared__ bf16_t Asb[2][64 * 40];   // 64 rows x 32 cols, ld=40 (80B rows, 16B-aligned)
  const int tid = threadIdx.x, wave = tid >> 5;
  const int lane = tid & 31, hf = lane >> 4, ln = lane & 15;
  const int colBlock = blockIdx.y * 128;

  // staging: each thread handles one row, 16 consecutive cols (4 float4 -> 2 uint4)
  const int sr = tid >> 1;             // 0..63
  const int sc = (tid & 1) * 16;       // 0 or 16
  const float* srow = seq + ((size_t)blockIdx.x * 64 + sr) * kD + sc;

  v8f acc[4][2];
  for (int rt = 0; rt < 4; ++rt) { acc[rt][0] = v8zero(); acc[rt][1] = v8zero(); }

  float4 st0 = reinterpret_cast<const float4*>(srow)[0];
  float4 st1 = reinterpret_cast<const float4*>(srow)[1];
  float4 st2 = reinterpret_cast<const float4*>(srow)[2];
  float4 st3 = reinterpret_cast<const float4*>(srow)[3];

  for (int it = 0; it < 16; ++it) {
    const int kk = it * 32;
    const int buf = it & 1;
    {
      union { bf16_t b[8]; uint4 u; } p0, p1;
      p0.b[0] = (bf16_t)st0.x; p0.b[1] = (bf16_t)st0.y; p0.b[2] = (bf16_t)st0.z; p0.b[3] = (bf16_t)st0.w;
      p0.b[4] = (bf16_t)st1.x; p0.b[5] = (bf16_t)st1.y; p0.b[6] = (bf16_t)st1.z; p0.b[7] = (bf16_t)st1.w;
      p1.b[0] = (bf16_t)st2.x; p1.b[1] = (bf16_t)st2.y; p1.b[2] = (bf16_t)st2.z; p1.b[3] = (bf16_t)st2.w;
      p1.b[4] = (bf16_t)st3.x; p1.b[5] = (bf16_t)st3.y; p1.b[6] = (bf16_t)st3.z; p1.b[7] = (bf16_t)st3.w;
      uint4* d = reinterpret_cast<uint4*>(&Asb[buf][sr * 40 + sc]);
      d[0] = p0.u; d[1] = p1.u;
    }
    __syncthreads();   // stores of buf visible; all waves done reading buf^1
    if (it + 1 < 16) {
      const float* nrow = srow + kk + 32;
      st0 = reinterpret_cast<const float4*>(nrow)[0];
      st1 = reinterpret_cast<const float4*>(nrow)[1];
      st2 = reinterpret_cast<const float4*>(nrow)[2];
      st3 = reinterpret_cast<const float4*>(nrow)[3];
    }
    v16bf b0 = load_frag_b(w1t + (size_t)(colBlock + wave * 32) * kD + kk, kD);
    v16bf b1f = load_frag_b(w1t + (size_t)(colBlock + wave * 32 + 16) * kD + kk, kD);
    for (int rt = 0; rt < 4; ++rt) {
      v16bf a = load_frag_a(&Asb[buf][rt * 16 * 40], 40);
      acc[rt][0] = wmma_bf16(a, b0, acc[rt][0]);
      acc[rt][1] = wmma_bf16(a, b1f, acc[rt][1]);
    }
  }

  // bias + exact GELU, store bf16 to sh[f][b][col] (f = row 0..63, b = blockIdx.x)
  const int c0 = colBlock + wave * 32 + ln, c1 = c0 + 16;
  const float bv0 = b1[c0], bv1 = b1[c1];
  for (int rt = 0; rt < 4; ++rt) {
    for (int i = 0; i < 8; ++i) {
      int f = rt * 16 + i + 8 * hf;
      bf16_t* dst = sh + ((size_t)f * kB + blockIdx.x) * kD;
      float v0 = acc[rt][0][i] + bv0;
      float v1 = acc[rt][1][i] + bv1;
      dst[c0] = (bf16_t)(0.5f * v0 * (1.f + erff(v0 * 0.70710678118654752f)));
      dst[c1] = (bf16_t)(0.5f * v1 * (1.f + erff(v1 * 0.70710678118654752f)));
    }
  }
}

// ---------------- kernel 1b: in-place per-row LayerNorm over sh (131072 rows x 512) ----
// One wave per row: lane holds 16 contiguous elems (2 uint4), full-wave butterfly stats.
__global__ void __launch_bounds__(256)
k_ln(bf16_t* __restrict__ sh, const float* __restrict__ ln_g,
     const float* __restrict__ ln_b) {
  const int wave = threadIdx.x >> 5, lane = threadIdx.x & 31;
  bf16_t* p = sh + ((size_t)blockIdx.x * 8 + wave) * kD + lane * 16;
  union { uint4 q[2]; bf16_t h[16]; } d;
  d.q[0] = reinterpret_cast<const uint4*>(p)[0];
  d.q[1] = reinterpret_cast<const uint4*>(p)[1];
  float x[16];
  float s = 0.f, q2 = 0.f;
  for (int i = 0; i < 16; ++i) { x[i] = (float)d.h[i]; s += x[i]; q2 += x[i] * x[i]; }
  for (int m = 1; m < 32; m <<= 1) { s += __shfl_xor(s, m, 32); q2 += __shfl_xor(q2, m, 32); }
  float mu = s * (1.f / 512.f);
  float rstd = rsqrtf(q2 * (1.f / 512.f) - mu * mu + 1e-5f);
  const float* gp = ln_g + lane * 16;
  const float* bp = ln_b + lane * 16;
  for (int i = 0; i < 16; ++i) {
    float y = (x[i] - mu) * rstd * gp[i] + bp[i];
    d.h[i] = (bf16_t)y;
  }
  reinterpret_cast<uint4*>(p)[0] = d.q[0];
  reinterpret_cast<uint4*>(p)[1] = d.q[1];
}

// ---------------- kernel 2: h1[f] = relu(shared[f] @ Wh1[f] + bias), bf16 ----------------
// Grid (B/64, 2, 64). Block 128 = 4 waves, each wave: 64 rows x 32 cols (B reused 4x).
__global__ void __launch_bounds__(128, 1)
k_head1(const bf16_t* __restrict__ sh, const bf16_t* __restrict__ wh1t,
        const float* __restrict__ bc1, const float* __restrict__ bn1,
        const float* __restrict__ bb1, bf16_t* __restrict__ h1) {
  const int tid = threadIdx.x, wave = tid >> 5;
  const int lane = tid & 31, hf = lane >> 4, ln = lane & 15;
  const int f = blockIdx.z;
  const int rowBase = blockIdx.x * 64;
  const int colBase = blockIdx.y * 128 + wave * 32;

  const bf16_t* A  = sh + ((size_t)f * kB + rowBase) * kD;
  const bf16_t* Wt = wh1t + (size_t)f * kH * kD;
  const float* bias = (f < 32) ? (bc1 + (size_t)f * kH)
                    : (f < 48) ? (bn1 + (size_t)(f - 32) * kH)
                               : (bb1 + (size_t)(f - 48) * kH);

  v8f acc[4][2];
  for (int rt = 0; rt < 4; ++rt) { acc[rt][0] = v8zero(); acc[rt][1] = v8zero(); }

  for (int kk = 0; kk < kD; kk += 32) {
    v16bf b0 = load_frag_b(Wt + (size_t)colBase * kD + kk, kD);
    v16bf b1f = load_frag_b(Wt + (size_t)(colBase + 16) * kD + kk, kD);
    for (int rt = 0; rt < 4; ++rt) {
      v16bf a = load_frag_a(A + (size_t)(rt * 16) * kD + kk, kD);
      acc[rt][0] = wmma_bf16(a, b0, acc[rt][0]);
      acc[rt][1] = wmma_bf16(a, b1f, acc[rt][1]);
    }
  }
  bf16_t* dst = h1 + ((size_t)f * kB + rowBase) * kH;
  const int c0 = colBase + ln, c1 = c0 + 16;
  const float bv0 = bias[c0], bv1 = bias[c1];
  for (int rt = 0; rt < 4; ++rt) {
    for (int i = 0; i < 8; ++i) {
      int row = rt * 16 + i + 8 * hf;
      dst[(size_t)row * kH + c0] = (bf16_t)fmaxf(acc[rt][0][i] + bv0, 0.f);
      dst[(size_t)row * kH + c1] = (bf16_t)fmaxf(acc[rt][1][i] + bv1, 0.f);
    }
  }
}

// ---------------- kernel 3: categorical layer-2 + fused online-softmax CE ----------------
// Grid (32, B/32). Block 128 = 4 waves; 32 rows staged in LDS (B frags reused 2x).
// Lane-local online softmax over each lane's private column subset; one merge at end.
__global__ void __launch_bounds__(128, 2)
k_cat(const bf16_t* __restrict__ h1, const bf16_t* __restrict__ wc2t,
      const float* __restrict__ bc2, const int* __restrict__ cat_t,
      const unsigned char* __restrict__ mask, float* __restrict__ loss_acc) {
  __shared__ bf16_t Asm[32 * kLdA];
  __shared__ float wm[4][32], wsum[4][32], wtg[4][32];
  const int tid = threadIdx.x, wave = tid >> 5;
  const int lane = tid & 31, hf = lane >> 4, ln = lane & 15;
  const int f = blockIdx.x;
  const int rowBase = blockIdx.y * 32;

  const bf16_t* A = h1 + ((size_t)f * kB + rowBase) * kH;
  for (int i = tid; i < 1024; i += 128) {
    int r = i >> 5, c = (i & 31) * 8;
    *reinterpret_cast<uint4*>(&Asm[r * kLdA + c]) =
        *reinterpret_cast<const uint4*>(&A[(size_t)r * kH + c]);
  }
  __syncthreads();

  int tcol[2][8];
  for (int rt = 0; rt < 2; ++rt)
    for (int i = 0; i < 8; ++i)
      tcol[rt][i] = cat_t[(size_t)(rowBase + rt * 16 + i + 8 * hf) * kNC + f];

  float m_run[2][8], s_run[2][8], t_acc[2][8];
  for (int rt = 0; rt < 2; ++rt)
    for (int i = 0; i < 8; ++i) { m_run[rt][i] = NEG_BIG; s_run[rt][i] = 0.f; t_acc[rt][i] = 0.f; }

  const bf16_t* Wt = wc2t + (size_t)f * kVp * kH;
  for (int chunk = wave; chunk < 63; chunk += 4) {
    int colBase = chunk * 16;
    v8f acc0 = v8zero(), acc1 = v8zero();
    for (int kc = 0; kc < 8; ++kc) {
      v16bf b = load_frag_b(Wt + (size_t)colBase * kH + kc * 32, kH);
      v16bf a0 = load_frag_a(&Asm[0 * 16 * kLdA + kc * 32], kLdA);
      v16bf a1 = load_frag_a(&Asm[1 * 16 * kLdA + kc * 32], kLdA);
      acc0 = wmma_bf16(a0, b, acc0);
      acc1 = wmma_bf16(a1, b, acc1);
    }
    int col = colBase + ln;
    if (col < kV) {
      float bias = bc2[(size_t)f * kV + col];
      for (int rt = 0; rt < 2; ++rt) {
        for (int i = 0; i < 8; ++i) {
          float val = (rt ? acc1[i] : acc0[i]) + bias;
          if (val > m_run[rt][i]) {
            s_run[rt][i] = s_run[rt][i] * __expf(m_run[rt][i] - val) + 1.f;
            m_run[rt][i] = val;
          } else {
            s_run[rt][i] += __expf(val - m_run[rt][i]);
          }
          if (col == tcol[rt][i]) t_acc[rt][i] += val;
        }
      }
    }
  }
  for (int rt = 0; rt < 2; ++rt)
    for (int i = 0; i < 8; ++i) {
      float mm = m_run[rt][i], ss = s_run[rt][i], tt = t_acc[rt][i];
      for (int m = 1; m < 16; m <<= 1) {
        float mo = __shfl_xor(mm, m, 32);
        float so = __shfl_xor(ss, m, 32);
        float mn = fmaxf(mm, mo);
        ss = ss * __expf(mm - mn) + so * __expf(mo - mn);
        mm = mn;
        tt += __shfl_xor(tt, m, 32);
      }
      if (ln == 0) {
        int row = rt * 16 + i + 8 * hf;
        wm[wave][row] = mm; wsum[wave][row] = ss; wtg[wave][row] = tt;
      }
    }
  __syncthreads();
  if (tid < 32) {
    int row = tid;
    float gm = NEG_BIG;
    for (int w = 0; w < 4; ++w) gm = fmaxf(gm, wm[w][row]);
    float gs = 0.f, tl = 0.f;
    for (int w = 0; w < 4; ++w) {
      gs += wsum[w][row] * __expf(wm[w][row] - gm);
      tl += wtg[w][row];            // exactly one wave/lane contributed the target logit
    }
    float ce = gm + __logf(gs) - tl;
    int b = rowBase + row;
    if (mask[(size_t)b * kNF + f]) atomicAdd(&loss_acc[f], ce);
  }
}

// ---------------- kernel 4: numeric layer-2 + fused MSE ----------------
// Grid (16, B/32). Block 128 = 4 waves; 32 rows staged in LDS; wave w: cols w*128..+127.
__global__ void __launch_bounds__(128, 2)
k_num(const bf16_t* __restrict__ h1, const bf16_t* __restrict__ wn2t,
      const float* __restrict__ bn2, const float* __restrict__ targets,
      const unsigned char* __restrict__ mask, float* __restrict__ loss_acc) {
  __shared__ bf16_t Asm[32 * kLdA];
  __shared__ float redN[4][32];
  const int tid = threadIdx.x, wave = tid >> 5;
  const int lane = tid & 31, hf = lane >> 4, ln = lane & 15;
  const int fh = blockIdx.x;        // 0..15
  const int fg = 32 + fh;           // global feature index
  const int rowBase = blockIdx.y * 32;

  const bf16_t* A = h1 + ((size_t)fg * kB + rowBase) * kH;
  for (int i = tid; i < 1024; i += 128) {
    int r = i >> 5, c = (i & 31) * 8;
    *reinterpret_cast<uint4*>(&Asm[r * kLdA + c]) =
        *reinterpret_cast<const uint4*>(&A[(size_t)r * kH + c]);
  }
  __syncthreads();

  float sq_acc[2][8];
  for (int rt = 0; rt < 2; ++rt)
    for (int i = 0; i < 8; ++i) sq_acc[rt][i] = 0.f;

  const bf16_t* Wt = wn2t + (size_t)fh * kD * kH;
  for (int t = 0; t < 8; ++t) {
    int colBase = wave * 128 + t * 16;
    v8f acc0 = v8zero(), acc1 = v8zero();
    for (int kc = 0; kc < 8; ++kc) {
      v16bf b = load_frag_b(Wt + (size_t)colBase * kH + kc * 32, kH);
      v16bf a0 = load_frag_a(&Asm[0 * 16 * kLdA + kc * 32], kLdA);
      v16bf a1 = load_frag_a(&Asm[1 * 16 * kLdA + kc * 32], kLdA);
      acc0 = wmma_bf16(a0, b, acc0);
      acc1 = wmma_bf16(a1, b, acc1);
    }
    int col = colBase + ln;
    float bias = bn2[(size_t)fh * kD + col];
    for (int rt = 0; rt < 2; ++rt) {
      for (int i = 0; i < 8; ++i) {
        int b = rowBase + rt * 16 + i + 8 * hf;
        float pred = (rt ? acc1[i] : acc0[i]) + bias;
        float tgt = targets[((size_t)b * kNF + fg) * kD + col];
        float d = pred - tgt;
        sq_acc[rt][i] += d * d;
      }
    }
  }
  for (int rt = 0; rt < 2; ++rt)
    for (int i = 0; i < 8; ++i) {
      float sq = sq_acc[rt][i];
      for (int m = 1; m < 16; m <<= 1) sq += __shfl_xor(sq, m, 32);
      if (ln == 0) redN[wave][rt * 16 + i + 8 * hf] = sq;
    }
  __syncthreads();
  if (tid < 32) {
    float s = 0.f;
    for (int w = 0; w < 4; ++w) s += redN[w][tid];
    int b = rowBase + tid;
    float mse = s * (1.f / 512.f);
    if (mask[(size_t)b * kNF + fg]) atomicAdd(&loss_acc[fg], mse);
  }
}

// ---------------- kernel 5: bool heads (N=2, scalar) + CE ----------------
__global__ void k_bool(const bf16_t* __restrict__ h1, const bf16_t* __restrict__ wb2t,
                       const float* __restrict__ bb2, const int* __restrict__ bool_t,
                       const unsigned char* __restrict__ mask, float* __restrict__ loss_acc) {
  int idx = blockIdx.x * blockDim.x + threadIdx.x;
  if (idx >= kB * kNB) return;
  int b = idx >> 4, fh = idx & 15;
  int fg = 48 + fh;
  const bf16_t* hrow = h1 + ((size_t)fg * kB + b) * kH;
  const bf16_t* w0 = wb2t + ((size_t)fh * 2 + 0) * kH;
  const bf16_t* w1 = wb2t + ((size_t)fh * 2 + 1) * kH;
  float a0 = bb2[fh * 2 + 0], a1 = bb2[fh * 2 + 1];
  for (int k = 0; k < kH; ++k) {
    float h = (float)hrow[k];
    a0 += h * (float)w0[k];
    a1 += h * (float)w1[k];
  }
  float m = fmaxf(a0, a1);
  float lse = m + __logf(__expf(a0 - m) + __expf(a1 - m));
  int t = bool_t[(size_t)b * kNB + fh];
  float ce = lse - (t ? a1 : a0);
  if (mask[(size_t)b * kNF + fg]) atomicAdd(&loss_acc[fg], ce);
}

// ---------------- kernel 6: masked-mean denominators + total ----------------
__global__ void k_final(const float* __restrict__ loss_acc,
                        const unsigned char* __restrict__ mask, float* __restrict__ out) {
  __shared__ float part[64];
  int f = threadIdx.x;               // 64 threads
  int cnt = 0;
  for (int b = 0; b < kB; ++b) cnt += mask[(size_t)b * kNF + f] ? 1 : 0;
  float fl = loss_acc[f] / fmaxf((float)cnt, 1.f);
  part[f] = (cnt > 0) ? fl : 0.f;
  __syncthreads();
  if (f == 0) {
    float s = 0.f;
    for (int i = 0; i < 64; ++i) s += part[i];
    out[0] = s;
  }
}

// ---------------- launcher ----------------
extern "C" void kernel_launch(void* const* d_in, const int* in_sizes, int n_in,
                              void* d_out, int out_size, void* d_ws, size_t ws_size,
                              hipStream_t stream) {
  (void)in_sizes; (void)n_in; (void)out_size; (void)ws_size;
  const float* seq      = (const float*)d_in[0];
  const float* targets  = (const float*)d_in[1];
  const unsigned char* mask = (const unsigned char*)d_in[2]; // jnp bool (1 byte)
  const int* cat_t      = (const int*)d_in[3];
  const int* bool_t     = (const int*)d_in[4];
  const float* W1  = (const float*)d_in[5];
  const float* b1  = (const float*)d_in[6];
  const float* ln_g = (const float*)d_in[7];
  const float* ln_b = (const float*)d_in[8];
  const float* Wc1 = (const float*)d_in[9];
  const float* bc1 = (const float*)d_in[10];
  const float* Wc2 = (const float*)d_in[11];
  const float* bc2 = (const float*)d_in[12];
  const float* Wn1 = (const float*)d_in[13];
  const float* bn1 = (const float*)d_in[14];
  const float* Wn2 = (const float*)d_in[15];
  const float* bn2 = (const float*)d_in[16];
  const float* Wb1 = (const float*)d_in[17];
  const float* bb1 = (const float*)d_in[18];
  const float* Wb2 = (const float*)d_in[19];
  const float* bb2 = (const float*)d_in[20];

  // workspace carve-out (256B aligned)
  char* ws = (char*)d_ws;
  size_t off = 0;
  auto carve = [&](size_t bytes) -> char* {
    char* p = ws + off;
    off += (bytes + 255) & ~(size_t)255;
    return p;
  };
  bf16_t* w1t  = (bf16_t*)carve((size_t)kD * kD * 2);
  bf16_t* wh1t = (bf16_t*)carve((size_t)kNF * kH * kD * 2);
  bf16_t* wc2t = (bf16_t*)carve((size_t)kNC * kVp * kH * 2);
  bf16_t* wn2t = (bf16_t*)carve((size_t)kNN * kD * kH * 2);
  bf16_t* wb2t = (bf16_t*)carve((size_t)kNB * 2 * kH * 2);
  bf16_t* sh   = (bf16_t*)carve((size_t)kNF * kB * kD * 2);
  bf16_t* h1   = (bf16_t*)carve((size_t)kNF * kB * kH * 2);
  float* loss_acc = (float*)carve(64 * sizeof(float));

  k_prep<<<dim3(512), dim3(256), 0, stream>>>(W1, Wc1, Wc2, Wn1, Wn2, Wb1, Wb2,
                                              w1t, wh1t, wc2t, wn2t, wb2t, loss_acc);
  k_gemm1<<<dim3((kB * kNF) / 64, 4), dim3(128), 0, stream>>>(seq, w1t, b1, sh);
  k_ln<<<dim3((kB * kNF) / 8), dim3(256), 0, stream>>>(sh, ln_g, ln_b);
  k_head1<<<dim3(kB / 64, 2, kNF), dim3(128), 0, stream>>>(sh, wh1t, bc1, bn1, bb1, h1);
  k_cat<<<dim3(kNC, kB / 32), dim3(128), 0, stream>>>(h1, wc2t, bc2, cat_t, mask, loss_acc);
  k_num<<<dim3(kNN, kB / 32), dim3(128), 0, stream>>>(h1, wn2t, bn2, targets, mask, loss_acc);
  k_bool<<<dim3((kB * kNB + 255) / 256), dim3(256), 0, stream>>>(h1, wb2t, bb2, bool_t, mask, loss_acc);
  k_final<<<dim3(1), dim3(64), 0, stream>>>(loss_acc, mask, (float*)d_out);
}
